// GATLayer_22308060135710
// MI455X (gfx1250) — matrix operand order repaired and checked by reading, not compiled, not run on hardware
//
#include <hip/hip_runtime.h>

// ---------------------------------------------------------------------------
// GAT layer for MI455X (gfx1250, wave32, WMMA 16x16x32 f16)
// B=16, N=1024, IN_F=256, HEADS=4, OUT_F=64  (F = HEADS*OUT_F = 256)
// ---------------------------------------------------------------------------

typedef _Float16 half16 __attribute__((ext_vector_type(16)));
typedef _Float16 half8v __attribute__((ext_vector_type(8)));
typedef _Float16 half4v __attribute__((ext_vector_type(4)));
typedef float    float8 __attribute__((ext_vector_type(8)));

#define WMMA_F16(A, B, C) \
  __builtin_amdgcn_wmma_f32_16x16x32_f16(false, (A), false, (B), (short)0, (C), false, false)

#define GAT_B     16
#define GAT_N     1024
#define GAT_F     256      // IN_F == HEADS*OUT_F == 256
#define GAT_H     4
#define GAT_D     64
#define NEG_SLOPE 0.2f
#define MASK_VAL  (-9.0e15f)

// ---------------------------------------------------------------------------
// Kernel 1a: fp32 -> f16 convert of x, 4 elements/thread (b128 in, b64 out)
// ---------------------------------------------------------------------------
__global__ void gat_cvt_x(const float* __restrict__ x, _Float16* __restrict__ xh, int n4) {
  int i = blockIdx.x * blockDim.x + threadIdx.x;
  if (i < n4) {
    float4 v = *(const float4*)(x + (size_t)i * 4);
    half4v h;
    h[0] = (_Float16)v.x; h[1] = (_Float16)v.y;
    h[2] = (_Float16)v.z; h[3] = (_Float16)v.w;
    *(half4v*)(xh + (size_t)i * 4) = h;
  }
}

// ---------------------------------------------------------------------------
// Kernel 1b: W (256x256) -> Wt f16 transposed: wt[n*256+k] = W[k*256+n]
// (so WMMA B-tiles become contiguous 32B loads along K)
// ---------------------------------------------------------------------------
__global__ void gat_cvt_wt(const float* __restrict__ W, _Float16* __restrict__ wt) {
  int i = blockIdx.x * blockDim.x + threadIdx.x;
  if (i < GAT_F * GAT_F) {
    int nrow = i >> 8;
    int k    = i & 255;
    wt[i] = (_Float16)W[k * GAT_F + nrow];
  }
}

// ---------------------------------------------------------------------------
// Kernel 2: GEMM1  h = x @ W   (M=16384, K=256, N=256), f16 WMMA, f32 accum.
//   grid = 1024 blocks (one 16-row M tile each), 128 threads = 4 waves,
//   wave w computes N range [w*64, w*64+64) as 4 C tiles, K-loop of 8.
//   Software-pipelined: iteration kt+32's A/B loads are issued before
//   iteration kt's WMMAs, so no s_wait_loadcnt 0 sits in front of the XDL
//   work. The K-loop is fully unrolled so the buffer rotation is free.
// Outputs: h fp32 [b,node,feat]  and  ht f16 [b,feat,node] (for GEMM2 B).
// ---------------------------------------------------------------------------
__global__ void gat_gemm1(const _Float16* __restrict__ xh,
                          const _Float16* __restrict__ wt,
                          float* __restrict__ h,
                          _Float16* __restrict__ ht) {
  const int mtile = blockIdx.x;            // 0..1023
  const int wave  = threadIdx.x >> 5;      // 0..3
  const int lane  = threadIdx.x & 31;
  const int li    = lane & 15;
  const int half  = lane >> 4;
  const int Mrow  = mtile * 16 + li;       // A-matrix row owned by this lane

  const _Float16* xrow = xh + (size_t)Mrow * GAT_F;
  const _Float16* wrow = wt + (size_t)(wave * 64 + li) * GAT_F;  // +dt*16 rows

  float8 acc[4] = {};

  // Prologue: loads for kt = 0.
  half8v a0 = *(const half8v*)(xrow + half * 16);
  half8v a1 = *(const half8v*)(xrow + 8 + half * 16);
  half16 bm[4];
#pragma unroll
  for (int dt = 0; dt < 4; ++dt)
    bm[dt] = *(const half16*)(wrow + (size_t)(dt * 16) * GAT_F + half * 16);

#pragma unroll
  for (int kt = 0; kt < GAT_F; kt += 32) {
    // Issue next iteration's loads before this iteration's WMMAs.
    const int ktn = kt + 32;
    half8v a0n = {}, a1n = {};
    half16 bmn[4] = {};
    if (ktn < GAT_F) {
      a0n = *(const half8v*)(xrow + ktn + half * 16);
      a1n = *(const half8v*)(xrow + ktn + 8 + half * 16);
#pragma unroll
      for (int dt = 0; dt < 4; ++dt)
        bmn[dt] = *(const half16*)(wrow + (size_t)(dt * 16) * GAT_F + ktn + half * 16);
    }

    half16 amat;
#pragma unroll
    for (int t = 0; t < 8; ++t) { amat[t] = a0[t]; amat[8 + t] = a1[t]; }
#pragma unroll
    for (int dt = 0; dt < 4; ++dt) acc[dt] = WMMA_F16(amat, bm[dt], acc[dt]);

    // Rotate buffers (renamed away by full unroll).
    a0 = a0n; a1 = a1n;
#pragma unroll
    for (int dt = 0; dt < 4; ++dt) bm[dt] = bmn[dt];
  }

  // Epilogue: C layout — VGPR r holds row (r + 8*half), column li.
  const int b     = (mtile * 16) >> 10;
  const int node0 = (mtile * 16) & (GAT_N - 1);
#pragma unroll
  for (int dt = 0; dt < 4; ++dt) {
    const int feat = wave * 64 + dt * 16 + li;
#pragma unroll
    for (int r = 0; r < 8; ++r) {
      const int node = node0 + r + 8 * half;
      const float v  = acc[dt][r];
      h [((size_t)(b * GAT_N + node)) * GAT_F + feat] = v;
      ht[((size_t)(b * GAT_F + feat)) * GAT_N + node] = (_Float16)v;
    }
  }
}

// ---------------------------------------------------------------------------
// Kernel 3: attention scores, stored HEAD-MAJOR: s[b][head][n].
//   Head-major layout makes the per-lane s_dst runs in the attention kernel
//   contiguous (b128 loads) instead of 16 ds_bpermute per j-tile.
//   a is (128,4) row-major: a[d*4+head] (src), a[(64+d)*4+head] (dst)
// ---------------------------------------------------------------------------
__global__ void gat_scores(const float* __restrict__ h, const float* __restrict__ a,
                           float* __restrict__ ssrc, float* __restrict__ sdst) {
  int idx = blockIdx.x * blockDim.x + threadIdx.x;   // (b*N+n)*4 + head
  if (idx >= GAT_B * GAT_N * GAT_H) return;
  const int head = idx & 3;
  const int bn   = idx >> 2;
  const int b    = bn >> 10;
  const int node = bn & (GAT_N - 1);
  const float* hrow = h + (size_t)bn * GAT_F + head * GAT_D;
  float s0 = 0.f, s1 = 0.f;
#pragma unroll 8
  for (int d = 0; d < GAT_D; ++d) {
    const float hv = hrow[d];
    s0 += hv * a[d * GAT_H + head];
    s1 += hv * a[(GAT_D + d) * GAT_H + head];
  }
  const size_t o = ((size_t)(b * GAT_H + head)) * GAT_N + node;  // head-major
  ssrc[o] = s0;
  sdst[o] = s1;
}

// ---------------------------------------------------------------------------
// Kernel 4: fused masked-softmax attention + aggregation (flash style).
//   grid = B * N/16 = 1024 blocks; block = 128 threads = 4 waves = 4 heads.
//   Each wave: one 16-row i-tile of one head; streams j in tiles of 32.
//   Per j-tile: B tiles issued first (latency covered by score math),
//   scores -> online max/sum -> P (f16, A-layout) -> 4 WMMAs.
//   Accumulator rescale is skipped via uniform __any() when no row max grew.
// ---------------------------------------------------------------------------
__global__ void gat_attn(const _Float16* __restrict__ ht,
                         const float* __restrict__ ssrc,
                         const float* __restrict__ sdst,
                         const int* __restrict__ adj,
                         float* __restrict__ out) {
  const int b     = blockIdx.x >> 6;       // /64
  const int itile = blockIdx.x & 63;
  const int head  = threadIdx.x >> 5;
  const int lane  = threadIdx.x & 31;
  const int li    = lane & 15;
  const int half  = lane >> 4;
  const int i     = itile * 16 + li;       // P row owned by this lane

  const float* srow  = ssrc + ((size_t)(b * GAT_H + head)) * GAT_N;
  const float* sdrow = sdst + ((size_t)(b * GAT_H + head)) * GAT_N;
  const float  s_i   = srow[i];
  const int* adjrow  = adj + ((size_t)b * GAT_N + i) * GAT_N;
  const _Float16* htb = ht + ((size_t)(b * GAT_F + head * GAT_D)) * GAT_N;

  float8 acc[4] = {};
  float m_run = -3.0e38f;   // running row max
  float l_run = 0.f;        // running row denom

  for (int jt = 0; jt < GAT_N; jt += 32) {
    // Issue B tiles first: P(16x32) x H(32x16) operands, one 32B load each.
    half16 bm[4];
#pragma unroll
    for (int dt = 0; dt < 4; ++dt)
      bm[dt] = *(const half16*)(htb + (size_t)(dt * 16 + li) * GAT_N + jt + half * 16);

    // adjacency + s_dst for this lane's row: two 8-element runs each,
    // run offsets half*16 and half*16+8 match the B operand's contiguous k order.
    int   av[16];
    float sdv[16];
#pragma unroll
    for (int t = 0; t < 8; ++t) {
      av[t]      = adjrow[jt + half * 16 + t];
      av[8 + t]  = adjrow[jt + 8 + half * 16 + t];
      sdv[t]     = sdrow[jt + half * 16 + t];
      sdv[8 + t] = sdrow[jt + 8 + half * 16 + t];
    }
    if (jt + 32 < GAT_N) {
      __builtin_prefetch(adjrow + jt + 32, 0, 0);
      __builtin_prefetch(sdrow + jt + 32, 0, 0);
    }

    // masked leaky-relu scores (f16 A-layout element order)
    float sc[16];
#pragma unroll
    for (int e = 0; e < 16; ++e) {
      const float t  = s_i + sdv[e];
      const float lr = (t > 0.f) ? t : NEG_SLOPE * t;
      sc[e] = (av[e] > 0) ? lr : MASK_VAL;
    }

    // online softmax: row max across this lane + partner lane (xor 16)
    float mloc = sc[0];
#pragma unroll
    for (int e = 1; e < 16; ++e) mloc = fmaxf(mloc, sc[e]);
    const float mrow  = fmaxf(mloc, __shfl_xor(mloc, 16, 32));
    const bool  grew  = mrow > m_run;
    const float m_new = grew ? mrow : m_run;

    float psum = 0.f;
    half16 p16;
#pragma unroll
    for (int e = 0; e < 16; ++e) {
      const float p = __expf(sc[e] - m_new);
      psum += p;
      p16[e] = (_Float16)p;
    }
    psum += __shfl_xor(psum, 16, 32);

    // rescale only when some row's max actually grew (uniform branch; EXEC
    // stays all-ones around the WMMAs as required).
    if (__any(grew)) {
      const float scl = __expf(m_run - m_new);
      l_run = l_run * scl + psum;
#pragma unroll
      for (int r = 0; r < 8; ++r) {
        const float s_r = __shfl(scl, r + 8 * half, 32);  // C row r+8*half
#pragma unroll
        for (int dt = 0; dt < 4; ++dt) acc[dt][r] *= s_r;
      }
    } else {
      l_run += psum;
    }
    m_run = m_new;

#pragma unroll
    for (int dt = 0; dt < 4; ++dt) acc[dt] = WMMA_F16(p16, bm[dt], acc[dt]);
  }

  // normalize and store: out[b, node, head*64 + dt*16 + li]
  const int node0 = itile * 16;
#pragma unroll
  for (int r = 0; r < 8; ++r) {
    const float lsum = __shfl(l_run, r + 8 * half, 32);
    const float inv  = 1.0f / lsum;
    const int node   = node0 + r + 8 * half;
#pragma unroll
    for (int dt = 0; dt < 4; ++dt) {
      const int feat = head * GAT_D + dt * 16 + li;
      out[((size_t)(b * GAT_N + node)) * GAT_F + feat] = acc[dt][r] * inv;
    }
  }
}

// ---------------------------------------------------------------------------
// Host-side launch
// ---------------------------------------------------------------------------
extern "C" void kernel_launch(void* const* d_in, const int* in_sizes, int n_in,
                              void* d_out, int out_size, void* d_ws, size_t ws_size,
                              hipStream_t stream) {
  (void)in_sizes; (void)n_in; (void)out_size; (void)ws_size;

  const float* x   = (const float*)d_in[0];   // (16,1024,256)
  const int*   adj = (const int*)  d_in[1];   // (16,1024,1024)
  const float* W   = (const float*)d_in[2];   // (256,256)
  const float* a   = (const float*)d_in[3];   // (128,4)
  float* out = (float*)d_out;                 // (16,1024,256)

  // Workspace partition:
  //   xh   f16  16384*256   =  8 MiB
  //   wt   f16  256*256     = 128 KiB
  //   h    f32  16384*256   = 16 MiB
  //   ht   f16  16384*256   =  8 MiB
  //   ssrc f32  16*4*1024   = 256 KiB (head-major)
  //   sdst f32  16*4*1024   = 256 KiB (head-major)     total ~32.6 MiB
  char* ws = (char*)d_ws;
  _Float16* xh = (_Float16*)ws;                 ws += (size_t)GAT_B * GAT_N * GAT_F * 2;
  _Float16* wt = (_Float16*)ws;                 ws += (size_t)GAT_F * GAT_F * 2;
  float*    h  = (float*)ws;                    ws += (size_t)GAT_B * GAT_N * GAT_F * 4;
  _Float16* ht = (_Float16*)ws;                 ws += (size_t)GAT_B * GAT_N * GAT_F * 2;
  float* ssrc  = (float*)ws;                    ws += (size_t)GAT_B * GAT_N * GAT_H * 4;
  float* sdst  = (float*)ws;

  const int n4 = GAT_B * GAT_N * GAT_F / 4;                   // 1,048,576
  gat_cvt_x <<<(n4 + 255) / 256, 256, 0, stream>>>(x, xh, n4);
  gat_cvt_wt<<<(GAT_F * GAT_F + 255) / 256, 256, 0, stream>>>(W, wt);

  gat_gemm1<<<GAT_B * GAT_N / 16, 128, 0, stream>>>(xh, wt, h, ht);

  gat_scores<<<(GAT_B * GAT_N * GAT_H + 255) / 256, 256, 0, stream>>>(h, a, ssrc, sdst);

  gat_attn<<<GAT_B * (GAT_N / 16), 128, 0, stream>>>(ht, ssrc, sdst, adj, out);
}
